// EpipolarCrossAttentionLayer_15083925143760
// MI455X (gfx1250) — matrix operand order repaired and checked by reading, not compiled, not run on hardware
//
#include <hip/hip_runtime.h>
#include <stdint.h>

// ---------------------------------------------------------------------------
// EpipolarCrossAttentionLayer for MI455X (gfx1250, wave32, WMMA)
//   B=2, Q=2048, K=64, D_MODEL=256, H=8, Dk=32, FFN=1024
// Strategy: fuse K/V projection + softmax attention per (b,q) workgroup so the
// 256MB sampled_features tensor is read exactly once (non-temporal, single
// pass); all GEMMs via v_wmma_f32_16x16x32_bf16 with fp32 accumulation;
// LN/softmax/GELU in fp32.  k_attn LDS kept under 160KB so 2 workgroups
// co-reside per WGP and tile streaming overlaps WMMA compute.
// ---------------------------------------------------------------------------

typedef __bf16 v16bf __attribute__((ext_vector_type(16)));
typedef float  v8f   __attribute__((ext_vector_type(8)));
typedef float  f4    __attribute__((ext_vector_type(4)));
typedef unsigned short u16;
typedef unsigned int   u32;

#define DMODEL 256
#define NHEADS 8
#define DK     32
#define FFNDIM 1024
#define QLEN   2048
#define BATCH  2
#define KSAMP  64
#define ROWS   (BATCH * QLEN)   // 4096

// workspace byte offsets
#define OFF_WQ   (size_t)0          // 256x256 packed bf16 : 131072 B
#define OFF_WK   (size_t)131072
#define OFF_WV   (size_t)262144
#define OFF_WO   (size_t)393216
#define OFF_W1   (size_t)524288     // 256x1024 packed : 524288 B
#define OFF_W2   (size_t)1048576    // 1024x256 packed : 524288 B
#define OFF_QN   (size_t)1572864    // 4096x256 bf16   : 2 MB
#define OFF_Q    (size_t)3670016    // 4096x256 fp32   : 4 MB
#define OFF_AO   (size_t)7864320    // 4096x256 fp32   : 4 MB

union Frag {
  v16bf bf;
  uint4 u4[2];
};

__device__ __forceinline__ u16 f2bf(float f) {
  u32 u = __float_as_uint(f);
  u += 0x7FFFu + ((u >> 16) & 1u);   // round-to-nearest-even
  return (u16)(u >> 16);
}
__device__ __forceinline__ float bf2f(u16 h) {
  return __uint_as_float((u32)h << 16);
}
__device__ __forceinline__ u32 pack2(float a, float b) {
  return (u32)f2bf(a) | ((u32)f2bf(b) << 16);
}
__device__ __forceinline__ float wsum(float v) {
  #pragma unroll
  for (int m = 16; m >= 1; m >>= 1) v += __shfl_xor(v, m, 32);
  return v;
}
__device__ __forceinline__ float wmaxr(float v) {
  #pragma unroll
  for (int m = 16; m >= 1; m >>= 1) v = fmaxf(v, __shfl_xor(v, m, 32));
  return v;
}
__device__ __forceinline__ float gelu_exact(float v) {
  return 0.5f * v * (1.0f + erff(v * 0.70710678118654752440f));
}

// ---------------------------------------------------------------------------
// Core WMMA tile loop: C[64 x 8*16-strip] += A(64 x 32*KT bf16, LDS) * Wpacked
// A fragment (contiguous k order): lane(m=lane&15, hh=lane>>4):
//   halfs[0..15] = A[m][k0 + hh*16 + j]
// B fragment (pre-packed in global): lane(n=lane&15, hh=lane>>4) holds 16
//   contiguous K halfs of column n  -> one 32B contiguous load.
// ---------------------------------------------------------------------------
__device__ __forceinline__ void gemm_rowblock(
    const u16* __restrict__ A, int astride,
    const u16* __restrict__ Wpk, int KT,
    int mtile, int nbase, int lane, v8f acc[8])
{
  const int m  = mtile * 16 + (lane & 15);
  const int hh = lane >> 4;
  const uint4* wbase = (const uint4*)Wpk;
  for (int kt = 0; kt < KT; ++kt) {
    Frag a;
    const u16* ar = A + m * astride + kt * 32 + hh * 16;
    a.u4[0] = *(const uint4*)(ar);
    a.u4[1] = *(const uint4*)(ar + 8);
    #pragma unroll
    for (int i = 0; i < 8; ++i) {
      Frag b;
      const uint4* wp = wbase + ((size_t)((nbase + i) * KT + kt) * 32 + lane) * 2;
      b.u4[0] = wp[0];
      b.u4[1] = wp[1];
      acc[i] = __builtin_amdgcn_wmma_f32_16x16x32_bf16(
          false, a.bf, false, b.bf, (short)0, acc[i], false, false);
    }
  }
}

// ---------------------------------------------------------------------------
// Pack weight W (Kdim x N row-major fp32) into B-fragment order, bf16.
// grid = (N/16, Kdim/32), block = 32 threads (one wave).
// ---------------------------------------------------------------------------
__global__ void k_pack(const float* __restrict__ W, u16* __restrict__ dst, int N) {
  const int ntile = blockIdx.x;
  const int kt    = blockIdx.y;
  const int KT    = gridDim.y;
  const int lane  = threadIdx.x;
  const int n  = ntile * 16 + (lane & 15);
  const int k0 = kt * 32 + (lane >> 4) * 16;
  u32 p[8];
  #pragma unroll
  for (int j = 0; j < 8; ++j) {
    float a = W[(size_t)(k0 + 2 * j) * N + n];
    float b = W[(size_t)(k0 + 2 * j + 1) * N + n];
    p[j] = pack2(a, b);
  }
  uint4* out = (uint4*)dst + ((size_t)(ntile * KT + kt) * 32 + lane) * 2;
  uint4 lo; lo.x = p[0]; lo.y = p[1]; lo.z = p[2]; lo.w = p[3];
  uint4 hi; hi.x = p[4]; hi.y = p[5]; hi.z = p[6]; hi.w = p[7];
  out[0] = lo;
  out[1] = hi;
}

// ---------------------------------------------------------------------------
// LN1 over target_queries -> qn (bf16).  One wave per row; 512 blocks x 8 waves.
// ---------------------------------------------------------------------------
__global__ void __launch_bounds__(256) k_ln1(
    const float* __restrict__ tq, const float* __restrict__ g1,
    const float* __restrict__ be1, u16* __restrict__ qn)
{
  const int t = threadIdx.x, lane = t & 31, w = t >> 5;
  const int row = blockIdx.x * 8 + w;
  const float* xr = tq + (size_t)row * DMODEL + lane * 8;
  float4 a = *(const float4*)xr;
  float4 b = *(const float4*)(xr + 4);
  float xs[8] = {a.x, a.y, a.z, a.w, b.x, b.y, b.z, b.w};
  float s = 0.f;
  #pragma unroll
  for (int j = 0; j < 8; ++j) s += xs[j];
  const float mu = wsum(s) * (1.0f / 256.0f);
  float vs = 0.f;
  #pragma unroll
  for (int j = 0; j < 8; ++j) { float d = xs[j] - mu; vs += d * d; }
  const float var = wsum(vs) * (1.0f / 256.0f);
  const float rs = rsqrtf(var + 1e-5f);
  u32 p[4];
  #pragma unroll
  for (int j = 0; j < 4; ++j) {
    int c = lane * 8 + 2 * j;
    float y0 = (xs[2 * j] - mu) * rs * g1[c] + be1[c];
    float y1 = (xs[2 * j + 1] - mu) * rs * g1[c + 1] + be1[c + 1];
    p[j] = pack2(y0, y1);
  }
  uint4 u; u.x = p[0]; u.y = p[1]; u.z = p[2]; u.w = p[3];
  *(uint4*)(qn + (size_t)row * DMODEL + lane * 8) = u;
}

// ---------------------------------------------------------------------------
// Q projection: q = qn @ Wq + bq   (64 rows per block, WMMA)
// ---------------------------------------------------------------------------
__global__ void __launch_bounds__(256) k_qproj(
    const u16* __restrict__ qn, const u16* __restrict__ wqpk,
    const float* __restrict__ pbq, float* __restrict__ qout)
{
  extern __shared__ char smem[];
  u16* Al = (u16*)smem;                       // 64 x 264 bf16
  const int t = threadIdx.x, lane = t & 31, w = t >> 5;
  const int row0 = blockIdx.x * 64;

  const uint4* src4 = (const uint4*)(qn + (size_t)row0 * DMODEL);
  #pragma unroll
  for (int i = 0; i < 8; ++i) {
    int idx = t + i * 256;                    // 2048 uint4 total
    int row = idx >> 5, c8 = idx & 31;
    *(uint4*)(&Al[row * 264 + c8 * 8]) = src4[idx];
  }
  __syncthreads();

  const int mt = w & 3, nb = (w >> 2) * 8;
  v8f acc[8] = {};
  gemm_rowblock(Al, 264, wqpk, 8, mt, nb, lane, acc);

  const int r0 = mt * 16 + (lane >> 4) * 8;
  #pragma unroll
  for (int i = 0; i < 8; ++i) {
    int col = (nb + i) * 16 + (lane & 15);
    float bc = pbq[col];
    #pragma unroll
    for (int r = 0; r < 8; ++r)
      qout[(size_t)(row0 + r0 + r) * DMODEL + col] = acc[i][r] + bc;
  }
}

// ---------------------------------------------------------------------------
// Fused K/V projection + per-query attention.  One block per (b,q).
//   LDS: X bf16 (64x264) | K f32 (64x260) | V bf16 (64x264) | q | sc | wt
//   = 139264 B  ->  2 workgroups per 320KB WGP (load/compute overlap)
// ---------------------------------------------------------------------------
__global__ void __launch_bounds__(256) k_attn(
    const float* __restrict__ sf, const float* __restrict__ qf,
    const u16* __restrict__ wkpk, const u16* __restrict__ wvpk,
    const float* __restrict__ pbk, const float* __restrict__ pbv,
    float* __restrict__ attnw, float* __restrict__ ao)
{
  extern __shared__ char smem[];
  u16*   Xl = (u16*)smem;                     // 64 x 264 bf16      (33792)
  float* Kl = (float*)(smem + 33792);         // 64 x 260 fp32      (66560)
  u16*   Vl = (u16*)(smem + 100352);          // 64 x 264 bf16      (33792)
  float* qs = (float*)(smem + 134144);        // 256                 (1024)
  float* sc = (float*)(smem + 135168);        // 8 x 64              (2048)
  float* wt = (float*)(smem + 137216);        // 8 x 64              (2048)

  const int t = threadIdx.x, lane = t & 31, w = t >> 5;
  const int bq = blockIdx.x;
  const int b = bq >> 11, qi = bq & 2047;

  // phase 1: stage the 64x256 fp32 feature tile as bf16 in LDS.
  // Single-pass 256MB stream -> non-temporal loads (don't evict L2-resident
  // packed weights).
  const f4* src4 = (const f4*)(sf + (size_t)bq * (KSAMP * DMODEL));
  #pragma unroll
  for (int i = 0; i < 16; ++i) {
    int idx = t + i * 256;                    // 4096 float4 total
    f4 v = __builtin_nontemporal_load(&src4[idx]);
    int row = idx >> 6, c4 = idx & 63;
    uint2 p; p.x = pack2(v[0], v[1]); p.y = pack2(v[2], v[3]);
    *(uint2*)(&Xl[row * 264 + c4 * 4]) = p;
  }
  qs[t] = qf[(size_t)bq * DMODEL + t];
  __syncthreads();

  // phase 2: K = X@Wk + bk (fp32 in LDS), V = X@Wv + bv (bf16 in LDS)
  const int mt = w & 3, nb = (w >> 2) * 8;
  const int r0 = mt * 16 + (lane >> 4) * 8;
  {
    v8f acc[8] = {};
    gemm_rowblock(Xl, 264, wkpk, 8, mt, nb, lane, acc);
    #pragma unroll
    for (int i = 0; i < 8; ++i) {
      int col = (nb + i) * 16 + (lane & 15);
      float bc = pbk[col];
      #pragma unroll
      for (int r = 0; r < 8; ++r) Kl[(r0 + r) * 260 + col] = acc[i][r] + bc;
    }
  }
  {
    v8f acc[8] = {};
    gemm_rowblock(Xl, 264, wvpk, 8, mt, nb, lane, acc);
    #pragma unroll
    for (int i = 0; i < 8; ++i) {
      int col = (nb + i) * 16 + (lane & 15);
      float bc = pbv[col];
      #pragma unroll
      for (int r = 0; r < 8; ++r) Vl[(r0 + r) * 264 + col] = f2bf(acc[i][r] + bc);
    }
  }
  __syncthreads();

  // phase 3: scores (fp32) — 512 dot products of length 32
  const float scale = 0.17677669529663688f;   // 1/sqrt(32)
  #pragma unroll
  for (int p = 0; p < 2; ++p) {
    int idx = t + p * 256;
    int h = idx >> 6, kk = idx & 63;
    const float* qh = qs + h * DK;
    const float* kr = Kl + kk * 260 + h * DK;
    float s = 0.f;
    #pragma unroll
    for (int d = 0; d < DK; ++d) s += qh[d] * kr[d];
    sc[h * 64 + kk] = s * scale;
  }
  __syncthreads();

  // softmax: wave w owns head w (64 scores, 2 per lane)
  {
    float s0 = sc[w * 64 + lane], s1 = sc[w * 64 + lane + 32];
    float mx = wmaxr(fmaxf(s0, s1));
    float e0 = expf(s0 - mx), e1 = expf(s1 - mx);
    float inv = 1.0f / wsum(e0 + e1);
    float w0 = e0 * inv, w1 = e1 * inv;
    wt[w * 64 + lane] = w0; wt[w * 64 + lane + 32] = w1;
    size_t base = (((size_t)b * NHEADS + w) * QLEN + qi) * KSAMP;
    attnw[base + lane] = w0; attnw[base + lane + 32] = w1;
  }
  __syncthreads();

  // phase 4: attn_out = weights @ V
  {
    int h = t >> 5, d = t & 31;
    float o = 0.f;
    #pragma unroll
    for (int kk = 0; kk < KSAMP; ++kk)
      o += wt[h * 64 + kk] * bf2f(Vl[kk * 264 + h * DK + d]);
    ao[(size_t)bq * DMODEL + t] = o;
  }
}

// ---------------------------------------------------------------------------
// O projection + residual + LN2 + FFN(GELU) + residual.  64 rows per block.
//   LDS: bf16 buf (64x264) | x f32 (64x260) | h1 bf16 (64x1032)  = 232448 B
// ---------------------------------------------------------------------------
__global__ void __launch_bounds__(256) k_oproj_ffn(
    const float* __restrict__ ao, const float* __restrict__ tq,
    const u16* __restrict__ wopk, const float* __restrict__ pbo,
    const float* __restrict__ g2, const float* __restrict__ be2,
    const u16* __restrict__ w1pk, const float* __restrict__ pb1,
    const u16* __restrict__ w2pk, const float* __restrict__ pb2,
    float* __restrict__ out)
{
  extern __shared__ char smem[];
  u16*   Al = (u16*)smem;                     // 64 x 264 (attn bf16, then xn bf16)
  float* xf = (float*)(smem + 33792);         // 64 x 260
  u16*   h1 = (u16*)(smem + 100352);          // 64 x 1032

  const int t = threadIdx.x, lane = t & 31, w = t >> 5;
  const int row0 = blockIdx.x * 64;
  const int mt = w & 3, nb = (w >> 2) * 8;
  const int r0 = mt * 16 + (lane >> 4) * 8;

  // stage attn_out -> bf16 LDS
  const float4* src4 = (const float4*)(ao + (size_t)row0 * DMODEL);
  #pragma unroll
  for (int i = 0; i < 16; ++i) {
    int idx = t + i * 256;
    float4 v = src4[idx];
    int row = idx >> 6, c4 = idx & 63;
    uint2 p; p.x = pack2(v.x, v.y); p.y = pack2(v.z, v.w);
    *(uint2*)(&Al[row * 264 + c4 * 4]) = p;
  }
  __syncthreads();

  // x = tq + attn_out @ Wo + bo
  {
    v8f acc[8] = {};
    gemm_rowblock(Al, 264, wopk, 8, mt, nb, lane, acc);
    #pragma unroll
    for (int i = 0; i < 8; ++i) {
      int col = (nb + i) * 16 + (lane & 15);
      float bc = pbo[col];
      #pragma unroll
      for (int r = 0; r < 8; ++r) {
        int gr = row0 + r0 + r;
        xf[(r0 + r) * 260 + col] = tq[(size_t)gr * DMODEL + col] + acc[i][r] + bc;
      }
    }
  }
  __syncthreads();

  // LN2 -> xn bf16 (overwrites Al); wave w owns rows w*8 .. w*8+7
  for (int rr = 0; rr < 8; ++rr) {
    int row = w * 8 + rr;
    const float* xr = xf + row * 260 + lane * 8;
    float4 a = *(const float4*)xr;
    float4 bvv = *(const float4*)(xr + 4);
    float xs[8] = {a.x, a.y, a.z, a.w, bvv.x, bvv.y, bvv.z, bvv.w};
    float s = 0.f;
    #pragma unroll
    for (int j = 0; j < 8; ++j) s += xs[j];
    float mu = wsum(s) * (1.0f / 256.0f);
    float vs = 0.f;
    #pragma unroll
    for (int j = 0; j < 8; ++j) { float d = xs[j] - mu; vs += d * d; }
    float rs = rsqrtf(wsum(vs) * (1.0f / 256.0f) + 1e-5f);
    u32 p[4];
    #pragma unroll
    for (int j = 0; j < 4; ++j) {
      int c = lane * 8 + 2 * j;
      float y0 = (xs[2 * j] - mu) * rs * g2[c] + be2[c];
      float y1 = (xs[2 * j + 1] - mu) * rs * g2[c + 1] + be2[c + 1];
      p[j] = pack2(y0, y1);
    }
    uint4 u; u.x = p[0]; u.y = p[1]; u.z = p[2]; u.w = p[3];
    *(uint4*)(&Al[row * 264 + lane * 8]) = u;
  }
  __syncthreads();

  // h1 = gelu(xn @ W1 + b1), bf16 in LDS; N=1024 in 4 chunks of 256
  for (int ch = 0; ch < 4; ++ch) {
    v8f acc[8] = {};
    int nb1 = ch * 16 + (w >> 2) * 8;
    gemm_rowblock(Al, 264, w1pk, 8, mt, nb1, lane, acc);
    #pragma unroll
    for (int i = 0; i < 8; ++i) {
      int col = (nb1 + i) * 16 + (lane & 15);
      float bc = pb1[col];
      #pragma unroll
      for (int r = 0; r < 8; ++r)
        h1[(r0 + r) * 1032 + col] = f2bf(gelu_exact(acc[i][r] + bc));
    }
  }
  __syncthreads();

  // out = x + h1 @ W2 + b2   (K=1024 -> 32 k-steps)
  {
    v8f acc[8] = {};
    gemm_rowblock(h1, 1032, w2pk, 32, mt, nb, lane, acc);
    #pragma unroll
    for (int i = 0; i < 8; ++i) {
      int col = (nb + i) * 16 + (lane & 15);
      float bc = pb2[col];
      #pragma unroll
      for (int r = 0; r < 8; ++r) {
        int gr = row0 + r0 + r;
        out[(size_t)gr * DMODEL + col] = xf[(r0 + r) * 260 + col] + acc[i][r] + bc;
      }
    }
  }
}

// ---------------------------------------------------------------------------
extern "C" void kernel_launch(void* const* d_in, const int* in_sizes, int n_in,
                              void* d_out, int out_size, void* d_ws, size_t ws_size,
                              hipStream_t stream) {
  (void)in_sizes; (void)n_in; (void)out_size; (void)ws_size;

  const float* tq  = (const float*)d_in[0];
  const float* sf  = (const float*)d_in[1];
  const float* Wq  = (const float*)d_in[2];
  const float* bq  = (const float*)d_in[3];
  const float* Wk  = (const float*)d_in[4];
  const float* bk  = (const float*)d_in[5];
  const float* Wv  = (const float*)d_in[6];
  const float* bv  = (const float*)d_in[7];
  const float* Wo  = (const float*)d_in[8];
  const float* bo  = (const float*)d_in[9];
  const float* g1  = (const float*)d_in[10];
  const float* be1 = (const float*)d_in[11];
  const float* g2  = (const float*)d_in[12];
  const float* be2 = (const float*)d_in[13];
  const float* W1  = (const float*)d_in[14];
  const float* b1  = (const float*)d_in[15];
  const float* W2  = (const float*)d_in[16];
  const float* b2  = (const float*)d_in[17];

  char* ws = (char*)d_ws;
  u16* wqpk = (u16*)(ws + OFF_WQ);
  u16* wkpk = (u16*)(ws + OFF_WK);
  u16* wvpk = (u16*)(ws + OFF_WV);
  u16* wopk = (u16*)(ws + OFF_WO);
  u16* w1pk = (u16*)(ws + OFF_W1);
  u16* w2pk = (u16*)(ws + OFF_W2);
  u16*   qn = (u16*)(ws + OFF_QN);
  float* qf = (float*)(ws + OFF_Q);
  float* aoW = (float*)(ws + OFF_AO);

  float* out   = (float*)d_out;
  float* attnw = out + (size_t)ROWS * DMODEL;   // tuple: (out, attn_weights)

  // pack weights into WMMA B-fragment order (bf16)
  k_pack<<<dim3(16, 8),  32, 0, stream>>>(Wq, wqpk, 256);
  k_pack<<<dim3(16, 8),  32, 0, stream>>>(Wk, wkpk, 256);
  k_pack<<<dim3(16, 8),  32, 0, stream>>>(Wv, wvpk, 256);
  k_pack<<<dim3(16, 8),  32, 0, stream>>>(Wo, wopk, 256);
  k_pack<<<dim3(64, 8),  32, 0, stream>>>(W1, w1pk, 1024);
  k_pack<<<dim3(16, 32), 32, 0, stream>>>(W2, w2pk, 256);

  k_ln1  <<<512, 256, 0,      stream>>>(tq, g1, be1, qn);
  k_qproj<<<64,  256, 33792,  stream>>>(qn, wqpk, bq, qf);
  k_attn <<<ROWS, 256, 139264, stream>>>(sf, qf, wkpk, wvpk, bk, bv, attnw, aoW);
  k_oproj_ffn<<<64, 256, 232448, stream>>>(aoW, tq, wopk, bo, g2, be2,
                                           w1pk, b1, w2pk, b2, out);
}
